// HyperGraphBlock_11639361372556
// MI455X (gfx1250) — matrix operand order repaired and checked by reading, not compiled, not run on hardware
//
#include <hip/hip_runtime.h>
#include <hip/hip_bf16.h>

// ---------------------------------------------------------------------------
// HyperGraphBlock for MI455X (gfx1250, wave32, WMMA).
// B=8, N=2048, C_IN=768, C_OUT=159 (padded to 160), K=3.
// ---------------------------------------------------------------------------

#define Bn    8
#define Nn    2048
#define Rn    (Bn * Nn)        // 16384 rows total
#define Kc    768
#define CO    159
#define COP   160
#define EPSBN 1e-5f

typedef __attribute__((ext_vector_type(16))) __bf16 v16bf;
typedef __attribute__((ext_vector_type(8)))  __bf16 v8bf;
typedef __attribute__((ext_vector_type(8)))  float  v8f;

static __device__ __forceinline__ __bf16 f32_to_bf16(float f) {
  unsigned int u = __builtin_bit_cast(unsigned int, f);
  u += ((u >> 16) & 1u) + 0x7FFFu;            // round-to-nearest-even
  unsigned short h = (unsigned short)(u >> 16);
  return __builtin_bit_cast(__bf16, h);
}

static __device__ __forceinline__ v16bf make_afrag(const __bf16* ap) {
  // A fragment (16x32 bf16): elems 0..7 = ap[0..7], elems 8..15 = ap[16..23]
  v8bf lo = *(const v8bf*)(ap);
  v8bf hi = *(const v8bf*)(ap + 16);
  return __builtin_shufflevector(lo, hi, 0,1,2,3,4,5,6,7,8,9,10,11,12,13,14,15);
}

// Fully branch-free running top-2 (smallest) insertion.
// Distances: pure v_min_num/v_max_num ("2nd smallest = min(max(d,b0), b1)").
// Indices:   two flat (non-nested) selects -> v_cndmask each.
static __device__ __forceinline__ void top2_upd(float d, int id,
                                                float& b0, float& b1,
                                                int& i0, int& i1) {
  bool c0 = d < b0;
  bool c1 = d < b1;
  int t1 = c1 ? id : i1;       // candidate for slot1 when not displacing slot0
  i1 = c0 ? i0 : t1;
  i0 = c0 ? id : i0;
  b1 = fminf(fmaxf(d, b0), b1);
  b0 = fminf(d, b0);
}

// -------------------------------- utility ----------------------------------
__global__ void zero_f32(float* p, long n) {
  long i = (long)blockIdx.x * blockDim.x + threadIdx.x;
  if (i < n) p[i] = 0.0f;
}

// -------------------- convert x -> bf16 + row squared norms ----------------
__global__ __launch_bounds__(256) void convert_rows(const float* __restrict__ x,
                                                    __bf16* __restrict__ xb,
                                                    float* __restrict__ xsq) {
  int row = blockIdx.x;
  int t = threadIdx.x;
  const float* src = x + (size_t)row * Kc;
  __bf16* dst = xb + (size_t)row * Kc;
  float s = 0.0f;
  for (int k = t; k < Kc; k += 256) {
    float v = src[k];
    s += v * v;
    dst[k] = f32_to_bf16(v);
  }
  __shared__ float red[256];
  red[t] = s;
  __syncthreads();
  for (int off = 128; off > 0; off >>= 1) {
    if (t < off) red[t] += red[t + off];
    __syncthreads();
  }
  if (t == 0) xsq[row] = red[0];
}

// -------------------- theta (768x159) -> thetaT bf16 (160x768) -------------
__global__ __launch_bounds__(256) void convert_theta(const float* __restrict__ theta,
                                                     __bf16* __restrict__ thetaT) {
  int o = blockIdx.x;                 // 0..159
  for (int k = threadIdx.x; k < Kc; k += 256) {
    float v = (o < CO) ? theta[(size_t)k * CO + o] : 0.0f;
    thetaT[(size_t)o * Kc + k] = f32_to_bf16(v);
  }
}

// -------------------- kNN top-2 via bf16 WMMA Gram tiles -------------------
__global__ __launch_bounds__(256) void knn_top2(const __bf16* __restrict__ xb,
                                                const float* __restrict__ xsq,
                                                int* __restrict__ nn0,
                                                int* __restrict__ nn1) {
  __shared__ __align__(32) __bf16 ldsA[16 * Kc];   // 24 KB
  __shared__ float sxsq[16];
  __shared__ float cdist[16][128][2];              // 16 KB
  __shared__ int   cidx[16][128][2];               // 16 KB

  int tile = blockIdx.x;              // 0 .. B*(N/16)-1
  int b = tile >> 7;
  int rowBase = (tile & 127) << 4;
  size_t gRow = (size_t)b * Nn + rowBase;

  // stage 16 contiguous A rows into LDS with gfx1250 async global->LDS DMA.
  // 24576 bytes = 6 rounds x 256 lanes x b128.
  {
    unsigned ldsBase = (unsigned)(size_t)(void*)ldsA;
    const char* gsrc = (const char*)(xb + gRow * Kc);
#pragma unroll
    for (int r = 0; r < 6; r++) {
      int i = threadIdx.x + 256 * r;
      unsigned loff = ldsBase + i * 16;
      unsigned long long ga = (unsigned long long)(size_t)(gsrc + i * 16);
      asm volatile("global_load_async_to_lds_b128 %0, %1, off"
                   :: "v"(loff), "v"(ga) : "memory");
    }
    asm volatile("s_wait_asynccnt 0x0" ::: "memory");
  }
  if (threadIdx.x < 16) sxsq[threadIdx.x] = xsq[gRow + threadIdx.x];
  __syncthreads();

  int lane = threadIdx.x & 31;
  int wave = threadIdx.x >> 5;
  int half = lane >> 4;
  int l15  = lane & 15;

  float bd0[8], bd1[8];
  int   bi0[8], bi1[8];
#pragma unroll
  for (int v = 0; v < 8; v++) { bd0[v] = 3.4e38f; bd1[v] = 3.4e38f; bi0[v] = 0; bi1[v] = 0; }

  const __bf16* arow = ldsA + l15 * Kc + half * 8;

  // 16 column tiles per wave, processed in pairs: one A-fragment feeds 2 WMMAs
  for (int j = 0; j < 8; j++) {
    int ct0 = wave + 16 * j;
    int ct1 = ct0 + 8;
    int col0 = (ct0 << 4) + l15;
    int col1 = (ct1 << 4) + l15;
    const __bf16* brow0 = xb + ((size_t)b * Nn + col0) * Kc + half * 8;
    const __bf16* brow1 = xb + ((size_t)b * Nn + col1) * Kc + half * 8;
    if (j + 1 < 8) {
      __builtin_prefetch(xb + ((size_t)b * Nn + ((ct0 + 16) << 4) + l15) * Kc, 0, 1);
      __builtin_prefetch(xb + ((size_t)b * Nn + ((ct1 + 16) << 4) + l15) * Kc, 0, 1);
    }

    v8f acc0 = {}, acc1 = {};
#pragma unroll 2
    for (int k0 = 0; k0 < Kc; k0 += 32) {
      v16bf af = make_afrag(arow + k0);
      v16bf bf0 = make_afrag(brow0 + k0);
      v16bf bf1 = make_afrag(brow1 + k0);
      acc0 = __builtin_amdgcn_wmma_f32_16x16x32_bf16(
          false, af, false, bf0, (short)0, acc0, false, false);
      acc1 = __builtin_amdgcn_wmma_f32_16x16x32_bf16(
          false, af, false, bf1, (short)0, acc1, false, false);
    }
    float xc0 = xsq[(size_t)b * Nn + col0];
    float xc1 = xsq[(size_t)b * Nn + col1];
#pragma unroll
    for (int v = 0; v < 8; v++) {
      float rq = sxsq[v + 8 * half];
      float d0 = rq + xc0 - 2.0f * acc0[v];
      float d1 = rq + xc1 - 2.0f * acc1[v];
      top2_upd(d0, col0, bd0[v], bd1[v], bi0[v], bi1[v]);
      top2_upd(d1, col1, bd0[v], bd1[v], bi0[v], bi1[v]);
    }
  }

  int slot = wave * 16 + l15;
#pragma unroll
  for (int v = 0; v < 8; v++) {
    int row = v + 8 * half;
    cdist[row][slot][0] = bd0[v]; cidx[row][slot][0] = bi0[v];
    cdist[row][slot][1] = bd1[v]; cidx[row][slot][1] = bi1[v];
  }
  __syncthreads();

  if (threadIdx.x < 16) {
    int row = threadIdx.x;
    float d0 = 3.4e38f, d1 = 3.4e38f;
    int i0 = 0, i1 = 0;
    for (int s = 0; s < 128; s++)
      for (int j = 0; j < 2; j++)
        top2_upd(cdist[row][s][j], cidx[row][s][j], d0, d1, i0, i1);
    nn0[gRow + row] = i0;
    nn1[gRow + row] = i1;
  }
}

// -------------------- xt = x @ theta via bf16 WMMA -------------------------
__global__ __launch_bounds__(128) void gemm_xt(const __bf16* __restrict__ xb,
                                               const __bf16* __restrict__ thetaT,
                                               float* __restrict__ xt) {
  int wave = threadIdx.x >> 5;
  int lane = threadIdx.x & 31;
  int tile = blockIdx.x * 4 + wave;
  const int nTiles = (Rn / 16) * (COP / 16);   // 10240
  if (tile >= nTiles) return;
  int rt = tile / (COP / 16);
  int ctile = tile % (COP / 16);
  int rowBase = rt * 16;
  int half = lane >> 4;
  int l15 = lane & 15;

  const __bf16* ap = xb + (size_t)(rowBase + l15) * Kc + half * 8;
  const __bf16* bp = thetaT + (size_t)(ctile * 16 + l15) * Kc + half * 8;

  v8f acc = {};
#pragma unroll 4
  for (int k0 = 0; k0 < Kc; k0 += 32) {
    v16bf af = make_afrag(ap + k0);
    v16bf bfr = make_afrag(bp + k0);
    acc = __builtin_amdgcn_wmma_f32_16x16x32_bf16(
        false, af, false, bfr, (short)0, acc, false, false);
  }
  int col = ctile * 16 + l15;
#pragma unroll
  for (int v = 0; v < 8; v++)
    xt[(size_t)(rowBase + v + 8 * half) * COP + col] = acc[v];
}

// -------------------- hyperedge/node degrees -------------------------------
__global__ void degrees(const int* __restrict__ nn0, const int* __restrict__ nn1,
                        float* __restrict__ de, float* __restrict__ dn) {
  int i = blockIdx.x * blockDim.x + threadIdx.x;
  if (i >= Rn) return;
  int b = i >> 11;
  int n = i & (Nn - 1);
  int m0 = nn0[i], m1 = nn1[i], m2 = n;
  bool inc1 = (m1 != m0);
  bool inc2 = (m2 != m0) && (m2 != m1);
  de[i] = 1.0f + (inc1 ? 1.0f : 0.0f) + (inc2 ? 1.0f : 0.0f);
  size_t base = (size_t)b * Nn;
  atomicAdd(&dn[base + m0], 1.0f);
  if (inc1) atomicAdd(&dn[base + m1], 1.0f);
  if (inc2) atomicAdd(&dn[base + m2], 1.0f);
}

// -------------------- hyperedge features e = (1/de) H^T xt -----------------
__global__ __launch_bounds__(COP) void edge_feat(const int* __restrict__ nn0,
                                                 const int* __restrict__ nn1,
                                                 const float* __restrict__ de,
                                                 const float* __restrict__ xt,
                                                 float* __restrict__ e) {
  int i = blockIdx.x;          // hyperedge (global)
  int c = threadIdx.x;         // channel 0..159
  int b = i >> 11;
  int n = i & (Nn - 1);
  int m0 = nn0[i], m1 = nn1[i], m2 = n;
  bool inc1 = (m1 != m0);
  bool inc2 = (m2 != m0) && (m2 != m1);
  size_t base = (size_t)b * Nn;
  float v = xt[(base + m0) * COP + c];
  if (inc1) v += xt[(base + m1) * COP + c];
  if (inc2) v += xt[(base + m2) * COP + c];
  e[(size_t)i * COP + c] = v / de[i];
}

// -------------------- node scatter: acc += H e -----------------------------
__global__ __launch_bounds__(COP) void node_scatter(const int* __restrict__ nn0,
                                                    const int* __restrict__ nn1,
                                                    const float* __restrict__ e,
                                                    float* __restrict__ acc) {
  int i = blockIdx.x;
  int c = threadIdx.x;
  int b = i >> 11;
  int n = i & (Nn - 1);
  int m0 = nn0[i], m1 = nn1[i], m2 = n;
  bool inc1 = (m1 != m0);
  bool inc2 = (m2 != m0) && (m2 != m1);
  size_t base = (size_t)b * Nn;
  float v = e[(size_t)i * COP + c];
  atomicAdd(&acc[(base + m0) * COP + c], v);
  if (inc1) atomicAdd(&acc[(base + m1) * COP + c], v);
  if (inc2) atomicAdd(&acc[(base + m2) * COP + c], v);
}

// -------------------- per-channel stats (mean/var prep) --------------------
__global__ __launch_bounds__(256) void stats(float* __restrict__ acc,
                                             const float* __restrict__ dn,
                                             const float* __restrict__ bias,
                                             float* __restrict__ chs,
                                             float* __restrict__ chq) {
  int c = blockIdx.x;                       // 0..158 (real channels only)
  int r = blockIdx.y * 256 + threadIdx.x;   // 0..16383
  int t = threadIdx.x;
  float v = acc[(size_t)r * COP + c] / dn[r] + bias[c];
  acc[(size_t)r * COP + c] = v;
  __shared__ float s1[256], s2[256];
  s1[t] = v;
  s2[t] = v * v;
  __syncthreads();
  for (int off = 128; off > 0; off >>= 1) {
    if (t < off) { s1[t] += s1[t + off]; s2[t] += s2[t + off]; }
    __syncthreads();
  }
  if (t == 0) {
    atomicAdd(&chs[c], s1[0]);
    atomicAdd(&chq[c], s2[0]);
  }
}

// -------------------- BN + ReLU ---------------------------------------------
__global__ __launch_bounds__(COP + 1) void bn_relu(const float* __restrict__ acc,
                                                   const float* __restrict__ chs,
                                                   const float* __restrict__ chq,
                                                   const float* __restrict__ gamma,
                                                   const float* __restrict__ beta,
                                                   float* __restrict__ out) {
  int r = blockIdx.x;
  int c = threadIdx.x;
  if (c >= CO) return;
  const float invn = 1.0f / (float)Rn;
  float mean = chs[c] * invn;
  float var = chq[c] * invn - mean * mean;
  float v = (acc[(size_t)r * COP + c] - mean) * rsqrtf(var + EPSBN) * gamma[c] + beta[c];
  out[(size_t)r * CO + c] = v > 0.0f ? v : 0.0f;
}

// ---------------------------------------------------------------------------
extern "C" void kernel_launch(void* const* d_in, const int* in_sizes, int n_in,
                              void* d_out, int out_size, void* d_ws, size_t ws_size,
                              hipStream_t stream) {
  const float* x     = (const float*)d_in[0];
  const float* theta = (const float*)d_in[1];
  const float* bias  = (const float*)d_in[2];
  const float* gamma = (const float*)d_in[3];
  const float* beta  = (const float*)d_in[4];
  float* out = (float*)d_out;

  char* ws = (char*)d_ws;
  size_t o = 0;
  auto take = [&](size_t bytes) { size_t cur = o; o += (bytes + 255) & ~(size_t)255; return cur; };
  __bf16* xb     = (__bf16*)(ws + take((size_t)Rn * Kc * 2));
  __bf16* thetaT = (__bf16*)(ws + take((size_t)COP * Kc * 2));
  float*  xsq    = (float*)(ws + take((size_t)Rn * 4));
  int*    nn0    = (int*)(ws + take((size_t)Rn * 4));
  int*    nn1    = (int*)(ws + take((size_t)Rn * 4));
  float*  de     = (float*)(ws + take((size_t)Rn * 4));
  float*  dn     = (float*)(ws + take((size_t)Rn * 4));
  float*  xt     = (float*)(ws + take((size_t)Rn * COP * 4));
  float*  e      = (float*)(ws + take((size_t)Rn * COP * 4));
  float*  acc    = (float*)(ws + take((size_t)Rn * COP * 4));
  float*  chs    = (float*)(ws + take((size_t)COP * 4));
  float*  chq    = (float*)(ws + take((size_t)COP * 4));

  // zero accumulators (every call; graph replays must be deterministic)
  {
    long n1 = (long)Rn;
    zero_f32<<<(unsigned)((n1 + 255) / 256), 256, 0, stream>>>(dn, n1);
    long n2 = (long)Rn * COP;
    zero_f32<<<(unsigned)((n2 + 255) / 256), 256, 0, stream>>>(acc, n2);
    zero_f32<<<1, 256, 0, stream>>>(chs, COP);
    zero_f32<<<1, 256, 0, stream>>>(chq, COP);
  }

  convert_rows<<<Rn, 256, 0, stream>>>(x, xb, xsq);
  convert_theta<<<COP, 256, 0, stream>>>(theta, thetaT);

  knn_top2<<<Bn * (Nn / 16), 256, 0, stream>>>(xb, xsq, nn0, nn1);
  degrees<<<(Rn + 255) / 256, 256, 0, stream>>>(nn0, nn1, de, dn);

  gemm_xt<<<((Rn / 16) * (COP / 16)) / 4, 128, 0, stream>>>(xb, thetaT, xt);

  edge_feat<<<Rn, COP, 0, stream>>>(nn0, nn1, de, xt, e);
  node_scatter<<<Rn, COP, 0, stream>>>(nn0, nn1, e, acc);

  stats<<<dim3(CO, Rn / 256), 256, 0, stream>>>(acc, dn, bias, chs, chq);
  bn_relu<<<Rn, COP + 1, 0, stream>>>(acc, chs, chq, gamma, beta, out);
}
